// SingleHeadQKV_34806414967365
// MI455X (gfx1250) — matrix operand order repaired and checked by reading, not compiled, not run on hardware
//
#include <hip/hip_runtime.h>
#include <hip/hip_bf16.h>

#define B_ 4
#define T_ 2048
#define C_ 1024
// log2(1 - 2^-5) = log2(0.96875)
#define LOG2G (-0.045804023f)
// ln(10000)/1024
#define LNTH_D (9.210340372f / 1024.0f)

typedef __bf16 bf16_t;
typedef __attribute__((ext_vector_type(16))) __bf16 v16bf;
typedef __attribute__((ext_vector_type(8)))  __bf16 v8bf;
typedef __attribute__((ext_vector_type(8)))  float   v8f;
typedef int b128v __attribute__((vector_size(4 * sizeof(int))));  // int4 payload

// ---- CDNA5 async global->LDS copy (ASYNCcnt path), with safe fallback ------
#if defined(__HIP_DEVICE_COMPILE__) && __has_builtin(__builtin_amdgcn_global_load_async_to_lds_b128)
#define ASYNC_LDS 1
#endif

static __device__ __forceinline__ void cp_async16(bf16_t* lds, const bf16_t* gptr) {
#ifdef ASYNC_LDS
  __builtin_amdgcn_global_load_async_to_lds_b128(
      (__attribute__((address_space(1))) b128v*)const_cast<bf16_t*>(gptr),
      (__attribute__((address_space(3))) b128v*)lds, 0, 0);
#else
  *(v8bf*)lds = *(const v8bf*)gptr;
#endif
}
static __device__ __forceinline__ void async_fence() {
#ifdef ASYNC_LDS
#if __has_builtin(__builtin_amdgcn_s_wait_asynccnt)
  __builtin_amdgcn_s_wait_asynccnt(0);
#else
  asm volatile("s_wait_asynccnt 0x0" ::: "memory");
#endif
#endif
}

// ---- WMMA fragment helpers (wave32, 16x16x32 bf16) -------------------------
// A fragment: lanes 0-15 row M=lr (k 0-15); lanes 16-31 row M=lr
// (k 16-31). Source is row-major with `stride` elements per row.
static __device__ __forceinline__ v16bf ld_afrag(const bf16_t* base, int row0,
                                                 int stride, int lr, int hi) {
  const bf16_t* p = base + (size_t)(row0 + lr) * stride + (hi ? 16 : 0);
  return *(const v16bf*)p;
}
// B fragment: lane holds column N=lr, 16 contiguous K values (lanes>=16 start
// at k=16). Source is "n-major": column n is contiguous over k with `stride`.
static __device__ __forceinline__ v16bf ld_bfrag(const bf16_t* base, int col0,
                                                 int stride, int lr, int hi) {
  return *(const v16bf*)(base + (size_t)(col0 + lr) * stride + (hi ? 16 : 0));
}
static __device__ __forceinline__ v8f wmma_bf16(v16bf a, v16bf b, v8f c) {
  return __builtin_amdgcn_wmma_f32_16x16x32_bf16(false, a, false, b,
                                                 (short)0, c, false, false);
}
// pack 8 fp32 -> v8bf so staging stores become one ds_store_b128
static __device__ __forceinline__ v8bf pack8(const float* src) {
  v8bf r;
#pragma unroll
  for (int i = 0; i < 8; i++) r[i] = (bf16_t)src[i];
  return r;
}

// ---------------------------------------------------------------------------
// Kernel 1: Q/K/V projections (8192x1024x1024 GEMM x3) + fused interleaved
// rope on Q,K, output packed to bf16.
// grid = (64 m-blocks, 48 = 3 mats * 16 n-blocks), 256 threads (8 waves).
// Workgroup tile 128(M) x 64(N); wave tile 32x32 (2x2 WMMA tiles).
// ---------------------------------------------------------------------------
__global__ void __launch_bounds__(256, 2)
proj_rope_kernel(const float* __restrict__ X,
                 const float* __restrict__ WQ, const float* __restrict__ WK,
                 const float* __restrict__ WV,
                 bf16_t* __restrict__ Qb, bf16_t* __restrict__ Kb,
                 bf16_t* __restrict__ Vb) {
  const int mblk = blockIdx.x;
  const int mat  = blockIdx.y >> 4;   // 0:Q 1:K 2:V
  const int nblk = blockIdx.y & 15;
  const float* W = (mat == 0) ? WQ : (mat == 1) ? WK : WV;
  bf16_t*    Out = (mat == 0) ? Qb : (mat == 1) ? Kb : Vb;
  const int m0 = mblk * 128, n0 = nblk * 64;

  __shared__ __align__(32) bf16_t sA[128 * 32];  // X tile, row-major
  __shared__ __align__(32) bf16_t sB[64 * 32];   // W tile, n-major [n][k]

  const int tid = threadIdx.x;
  const int lane = tid & 31, wave = tid >> 5;
  const int lr = lane & 15, hi = lane >> 4;
  const int wr = wave >> 1, wc = wave & 1;       // waves 4x2 over 128x64

  v8f acc[2][2] = {};

  for (int k0 = 0; k0 < C_; k0 += 32) {
    {  // stage A: 128x32 fp32 -> bf16, packed b128 stores
      int r = tid >> 1, c = (tid & 1) * 16;
      const float* src = X + (size_t)(m0 + r) * C_ + k0 + c;
      __builtin_prefetch(src + 32, 0, 1);
      *(v8bf*)(sA + r * 32 + c)     = pack8(src);
      *(v8bf*)(sA + r * 32 + c + 8) = pack8(src + 8);
    }
    {  // stage B transposed into n-major: 32x64 -> sB[n][k]
      int kk = tid >> 3, nn = (tid & 7) * 8;
      const float* src = W + (size_t)(k0 + kk) * C_ + n0 + nn;
      __builtin_prefetch(src + 32 * C_, 0, 1);
#pragma unroll
      for (int i = 0; i < 8; i++) sB[(nn + i) * 32 + kk] = (bf16_t)src[i];
    }
    __syncthreads();
    v16bf af[2], bfv[2];
#pragma unroll
    for (int mi = 0; mi < 2; mi++) af[mi] = ld_afrag(sA, wr * 32 + mi * 16, 32, lr, hi);
#pragma unroll
    for (int ni = 0; ni < 2; ni++) bfv[ni] = ld_bfrag(sB, wc * 32 + ni * 16, 32, lr, hi);
#pragma unroll
    for (int mi = 0; mi < 2; mi++)
#pragma unroll
      for (int ni = 0; ni < 2; ni++) acc[mi][ni] = wmma_bf16(af[mi], bfv[ni], acc[mi][ni]);
    __syncthreads();
  }

  // epilogue: rope (Q,K only) + bf16 store. C-frag: VGPR j -> row m0+..+j(+8 hi),
  // col = n-tile + lr. Adjacent columns live in adjacent lanes -> shfl_xor(1).
#pragma unroll
  for (int mi = 0; mi < 2; mi++) {
#pragma unroll
    for (int ni = 0; ni < 2; ni++) {
#pragma unroll
      for (int j = 0; j < 8; j++) {
        float f = acc[mi][ni][j];
        int m = m0 + wr * 32 + mi * 16 + j + (hi ? 8 : 0);  // global row (b*T+t)
        int c = n0 + wc * 32 + ni * 16 + lr;
        float res = f;
        if (mat < 2) {
          float partner = __shfl_xor(f, 1, 32);
          int t = m & (T_ - 1);
          float inv = __expf(-(float)(c & ~1) * LNTH_D);    // theta^(-2i/d)
          float sn, cs;
          __sincosf((float)t * inv, &sn, &cs);
          res = f * cs + ((c & 1) ? partner * sn : -partner * sn);
        }
        Out[(size_t)m * C_ + c] = (bf16_t)res;
      }
    }
  }
}

// ---------------------------------------------------------------------------
// Transposes: build column-major copies so B-operand fragments are
// contiguous-in-K loads. bf16 [B][T][C] -> [B][C][T].
// ---------------------------------------------------------------------------
__global__ void __launch_bounds__(256)
transpose_bf16_kernel(const bf16_t* __restrict__ src, bf16_t* __restrict__ dst) {
  __shared__ bf16_t tile[32][33];
  int b = blockIdx.z;
  int t0 = blockIdx.x * 32, c0 = blockIdx.y * 32;
  const bf16_t* s = src + (size_t)b * T_ * C_;
  bf16_t*       d = dst + (size_t)b * T_ * C_;
  for (int i = threadIdx.y; i < 32; i += 8)
    tile[i][threadIdx.x] = s[(size_t)(t0 + i) * C_ + c0 + threadIdx.x];
  __syncthreads();
  for (int i = threadIdx.y; i < 32; i += 8)
    d[(size_t)(c0 + i) * T_ + t0 + threadIdx.x] = tile[threadIdx.x][i];
}

// S_n fp32 [B][C][O] -> St bf16 [B][O][C]
__global__ void __launch_bounds__(256)
transpose_f32_bf16_kernel(const float* __restrict__ src, bf16_t* __restrict__ dst) {
  __shared__ float tile[32][33];
  int b = blockIdx.z;
  int r0 = blockIdx.x * 32, c0 = blockIdx.y * 32;
  const float* s = src + (size_t)b * C_ * C_;
  bf16_t*      d = dst + (size_t)b * C_ * C_;
  for (int i = threadIdx.y; i < 32; i += 8)
    tile[i][threadIdx.x] = s[(size_t)(r0 + i) * C_ + c0 + threadIdx.x];
  __syncthreads();
  for (int i = threadIdx.y; i < 32; i += 8)
    d[(size_t)(c0 + i) * C_ + r0 + threadIdx.x] = (bf16_t)tile[threadIdx.x][i];
}

// ---------------------------------------------------------------------------
// Kernel 2: retention. One workgroup per (batch, 64-row chunk of T).
// acc[4][8] = 64 rows x 128 channels per wave (waves split the 1024 channels,
// so each 32-channel GN group stays inside one wave). Cross-chunk Q@S_n first
// (scaled by gamma^(t+1)), then flash-style s-block loop: all 8 waves build
// the shared 64x128 decayed score tile P in LDS, then accumulate P@V from the
// transposed V. sQ/sK staging uses CDNA5 async global->LDS DMA when available.
// Fused group-norm + transposed output store.
// ---------------------------------------------------------------------------
__global__ void __launch_bounds__(256, 1)
retention_kernel(const bf16_t* __restrict__ Qb, const bf16_t* __restrict__ Kb,
                 const bf16_t* __restrict__ Vt, const bf16_t* __restrict__ St,
                 const float* __restrict__ gnw, const float* __restrict__ gnb,
                 float* __restrict__ out) {
  const int blk = blockIdx.x;
  const int b   = blk >> 5;
  const int t0  = (blk & 31) * 64;
  const bf16_t* Q   = Qb + (size_t)b * T_ * C_;
  const bf16_t* K   = Kb + (size_t)b * T_ * C_;
  const bf16_t* Vtb = Vt + (size_t)b * C_ * T_;
  const bf16_t* Stb = St + (size_t)b * C_ * C_;
  float*       outb = out + (size_t)b * T_ * C_;

  __shared__ __align__(32) bf16_t sQ[64 * 32];
  __shared__ __align__(32) bf16_t sK[128 * 32];
  __shared__ __align__(32) bf16_t sP[64 * 128];

  const int tid = threadIdx.x, lane = tid & 31, wave = tid >> 5;
  const int lr = lane & 15, hi = lane >> 4;
  const int cw = wave * 128;   // this wave's channel base
  const int qr = tid >> 2, qc = (tid & 3) * 8;   // sQ staging coords
  const int kr = tid >> 1, kc = (tid & 1) * 16;  // sK staging coords

  v8f acc[4][8] = {};

  // ---- cross chunk: acc = Q @ S_n (K-dim = c) ----
  for (int k0 = 0; k0 < C_; k0 += 32) {
    cp_async16(sQ + qr * 32 + qc, Q + (size_t)(t0 + qr) * C_ + k0 + qc);
    async_fence();
    __syncthreads();
    v16bf aq[4];
#pragma unroll
    for (int mt = 0; mt < 4; mt++) aq[mt] = ld_afrag(sQ, mt * 16, 32, lr, hi);
#pragma unroll
    for (int nt = 0; nt < 8; nt++) {
      const bf16_t* bp = Stb + (size_t)(cw + nt * 16 + lr) * C_ + k0 + (hi ? 16 : 0);
      __builtin_prefetch(bp + 32, 0, 1);
      v16bf bv = *(const v16bf*)bp;
#pragma unroll
      for (int mt = 0; mt < 4; mt++) acc[mt][nt] = wmma_bf16(aq[mt], bv, acc[mt][nt]);
    }
    __syncthreads();
  }
  // scale cross-chunk term by e_t = gamma^(t+1)
#pragma unroll
  for (int mt = 0; mt < 4; mt++)
#pragma unroll
    for (int j = 0; j < 8; j++) {
      int t = t0 + mt * 16 + j + (hi ? 8 : 0);
      float sc = exp2f((float)(t + 1) * LOG2G);
#pragma unroll
      for (int nt = 0; nt < 8; nt++) acc[mt][nt][j] *= sc;
    }

  // ---- inner chunk: causal decayed attention over s-blocks of 128 ----
  const int nsblk = (t0 >> 7) + 1;
  for (int sb = 0; sb < nsblk; sb++) {
    const int s0 = sb * 128;
    // phase 1: P[:, 16w..16w+16] = Q K^T for this wave's 16 s-columns
    v8f pacc[4] = {};
    for (int k0 = 0; k0 < C_; k0 += 32) {
      cp_async16(sQ + qr * 32 + qc, Q + (size_t)(t0 + qr) * C_ + k0 + qc);
      cp_async16(sK + kr * 32 + kc,     K + (size_t)(s0 + kr) * C_ + k0 + kc);
      cp_async16(sK + kr * 32 + kc + 8, K + (size_t)(s0 + kr) * C_ + k0 + kc + 8);
      async_fence();
      __syncthreads();
      v16bf bk = ld_bfrag(sK, wave * 16, 32, lr, hi);  // K rows are B columns
#pragma unroll
      for (int mt = 0; mt < 4; mt++) {
        v16bf aq = ld_afrag(sQ, mt * 16, 32, lr, hi);
        pacc[mt] = wmma_bf16(aq, bk, pacc[mt]);
      }
      __syncthreads();
    }
    // phase 2: decay mask gamma^(t-s), pack P to LDS as bf16
#pragma unroll
    for (int mt = 0; mt < 4; mt++)
#pragma unroll
      for (int j = 0; j < 8; j++) {
        int rloc = mt * 16 + j + (hi ? 8 : 0);
        int t = t0 + rloc;
        int s = s0 + wave * 16 + lr;
        float d = (t >= s) ? exp2f((float)(t - s) * LOG2G) : 0.0f;
        sP[rloc * 128 + wave * 16 + lr] = (bf16_t)(pacc[mt][j] * d);
      }
    __syncthreads();
    // phase 3: acc += P @ V for this wave's 128 channels
#pragma unroll
    for (int ks = 0; ks < 4; ks++) {
      v16bf ap[4];
#pragma unroll
      for (int mt = 0; mt < 4; mt++) ap[mt] = ld_afrag(sP + ks * 32, mt * 16, 128, lr, hi);
#pragma unroll
      for (int nt = 0; nt < 8; nt++) {
        const bf16_t* bp = Vtb + (size_t)(cw + nt * 16 + lr) * T_ + s0 + ks * 32 + (hi ? 16 : 0);
        v16bf bv = *(const v16bf*)bp;
#pragma unroll
        for (int mt = 0; mt < 4; mt++) acc[mt][nt] = wmma_bf16(ap[mt], bv, acc[mt][nt]);
      }
    }
    __syncthreads();
  }

  // ---- fused group-norm (32 groups of 32 ch; 4 groups per wave) + store ----
  // Output layout after swapaxes+reshape: out[b][c*T + t].
#pragma unroll
  for (int mt = 0; mt < 4; mt++) {
#pragma unroll
    for (int g = 0; g < 4; g++) {
#pragma unroll
      for (int j = 0; j < 8; j++) {
        float a0 = acc[mt][2 * g][j], a1 = acc[mt][2 * g + 1][j];
        float s  = a0 + a1;
        float ss = a0 * a0 + a1 * a1;
#pragma unroll
        for (int m = 1; m < 16; m <<= 1) {  // stays inside 16-lane half
          s  += __shfl_xor(s,  m, 32);
          ss += __shfl_xor(ss, m, 32);
        }
        float mean = s * (1.0f / 32.0f);
        float var  = ss * (1.0f / 32.0f) - mean * mean;
        float rstd = rsqrtf(var + 1e-6f);
        int t  = t0 + mt * 16 + j + (hi ? 8 : 0);
        int c0 = cw + 32 * g + lr;
        int c1 = c0 + 16;
        outb[(size_t)c0 * T_ + t] = (a0 - mean) * rstd * gnw[c0] + gnb[c0];
        outb[(size_t)c1 * T_ + t] = (a1 - mean) * rstd * gnw[c1] + gnb[c1];
      }
    }
  }
}

// ---------------------------------------------------------------------------
// Kernel 3: R_i = K^T (V * gamma^(T-1-t)) + gamma^T * S_n.
// Per batch 1024x1024x2048 GEMM; decay folded into LDS-staged A tile.
// grid = (8 m-blocks, 16 n-blocks, B). Wave tile 32x32.
// ---------------------------------------------------------------------------
__global__ void __launch_bounds__(256, 2)
rstate_kernel(const bf16_t* __restrict__ Kt, const bf16_t* __restrict__ Vt,
              const float* __restrict__ Sn, float* __restrict__ Rout) {
  const int b = blockIdx.z;
  const bf16_t* A  = Kt + (size_t)b * C_ * T_;   // [c][t]
  const bf16_t* Bv = Vt + (size_t)b * C_ * T_;   // [o][t]
  const float*  S  = Sn + (size_t)b * C_ * C_;
  float*        R  = Rout + (size_t)b * C_ * C_;
  const int m0 = blockIdx.x * 128, n0 = blockIdx.y * 64;

  __shared__ __align__(32) bf16_t sA[128 * 32];

  const int tid = threadIdx.x, lane = tid & 31, wave = tid >> 5;
  const int lr = lane & 15, hi = lane >> 4;
  const int wr = wave >> 1, wc = wave & 1;

  v8f acc[2][2] = {};

  for (int k0 = 0; k0 < T_; k0 += 32) {
    {  // stage + decay-scale A tile: sA[r][t'] = Kt * gamma^(T-1-t), b128 stores
      int r = tid >> 1, cc = (tid & 1) * 16;
      const bf16_t* src = A + (size_t)(m0 + r) * T_ + k0 + cc;
      __builtin_prefetch(src + 32, 0, 1);
      v8bf lo, hi8;
#pragma unroll
      for (int i = 0; i < 8; i++) {
        float d0 = exp2f((float)(T_ - 1 - (k0 + cc + i)) * LOG2G);
        float d1 = exp2f((float)(T_ - 1 - (k0 + cc + 8 + i)) * LOG2G);
        lo[i]  = (bf16_t)((float)src[i] * d0);
        hi8[i] = (bf16_t)((float)src[8 + i] * d1);
      }
      *(v8bf*)(sA + r * 32 + cc)     = lo;
      *(v8bf*)(sA + r * 32 + cc + 8) = hi8;
    }
    __syncthreads();
    v16bf af[2], bfv[2];
#pragma unroll
    for (int mi = 0; mi < 2; mi++) af[mi] = ld_afrag(sA, wr * 32 + mi * 16, 32, lr, hi);
#pragma unroll
    for (int ni = 0; ni < 2; ni++) {
      const bf16_t* bp = Bv + (size_t)(n0 + wc * 32 + ni * 16 + lr) * T_ + k0 + (hi ? 16 : 0);
      bfv[ni] = *(const v16bf*)bp;
    }
#pragma unroll
    for (int mi = 0; mi < 2; mi++)
#pragma unroll
      for (int ni = 0; ni < 2; ni++) acc[mi][ni] = wmma_bf16(af[mi], bfv[ni], acc[mi][ni]);
    __syncthreads();
  }

  const float gT = exp2f(2048.0f * LOG2G);  // gamma^T (~6e-29, still normal)
#pragma unroll
  for (int mi = 0; mi < 2; mi++)
#pragma unroll
    for (int ni = 0; ni < 2; ni++)
#pragma unroll
      for (int j = 0; j < 8; j++) {
        int c = m0 + wr * 32 + mi * 16 + j + (hi ? 8 : 0);
        int o = n0 + wc * 32 + ni * 16 + lr;
        R[(size_t)c * C_ + o] = acc[mi][ni][j] + gT * S[(size_t)c * C_ + o];
      }
}

// ---------------------------------------------------------------------------
extern "C" void kernel_launch(void* const* d_in, const int* in_sizes, int n_in,
                              void* d_out, int out_size, void* d_ws, size_t ws_size,
                              hipStream_t stream) {
  const float* X  = (const float*)d_in[0];
  const float* Sn = (const float*)d_in[1];
  const float* WQ = (const float*)d_in[2];
  const float* WK = (const float*)d_in[3];
  const float* WV = (const float*)d_in[4];
  const float* gw = (const float*)d_in[5];
  const float* gb = (const float*)d_in[6];
  float* out = (float*)d_out;

  const size_t nBTC = (size_t)B_ * T_ * C_;   // 8388608
  const size_t nBCC = (size_t)B_ * C_ * C_;   // 4194304
  char* ws = (char*)d_ws;
  bf16_t* Qb = (bf16_t*)ws; ws += nBTC * 2;
  bf16_t* Kb = (bf16_t*)ws; ws += nBTC * 2;
  bf16_t* Vb = (bf16_t*)ws; ws += nBTC * 2;
  bf16_t* Kt = (bf16_t*)ws; ws += nBTC * 2;
  bf16_t* Vt = (bf16_t*)ws; ws += nBTC * 2;
  bf16_t* St = (bf16_t*)ws; ws += nBCC * 2;

  proj_rope_kernel<<<dim3(64, 48), 256, 0, stream>>>(X, WQ, WK, WV, Qb, Kb, Vb);
  transpose_bf16_kernel<<<dim3(64, 32, 4), dim3(32, 8), 0, stream>>>(Kb, Kt);
  transpose_bf16_kernel<<<dim3(64, 32, 4), dim3(32, 8), 0, stream>>>(Vb, Vt);
  transpose_f32_bf16_kernel<<<dim3(32, 32, 4), dim3(32, 8), 0, stream>>>(Sn, St);
  retention_kernel<<<dim3(128), 256, 0, stream>>>(Qb, Kb, Vt, St, gw, gb, out);
  rstate_kernel<<<dim3(8, 16, 4), 256, 0, stream>>>(Kt, Vt, Sn, out + nBTC);
}
